// SparseAttention_27788438405182
// MI455X (gfx1250) — matrix operand order repaired and checked by reading, not compiled, not run on hardware
//
#include <hip/hip_runtime.h>

// Problem constants (fixed by the reference).
#define BDIM   2
#define SEQ    4096
#define DMODEL 512
#define NHEADS 8
#define DK     64
#define KDIM   512
#define MROWS  (BDIM * SEQ)        // 8192

typedef __attribute__((ext_vector_type(16))) _Float16 v16h;
typedef __attribute__((ext_vector_type(8)))  _Float16 v8h;
typedef __attribute__((ext_vector_type(4)))  _Float16 v4h;
typedef __attribute__((ext_vector_type(8)))  float    v8f;
typedef __attribute__((ext_vector_type(4)))  unsigned int u32x4;
typedef __attribute__((ext_vector_type(8)))  int          i32x8;
typedef __attribute__((ext_vector_type(4)))  int          i32x4;

typedef __attribute__((address_space(1))) i32x4       gas_v4i; // global int4
typedef __attribute__((address_space(3))) i32x4       las_v4i; // LDS int4
typedef __attribute__((address_space(3))) _Float16    las_h;   // LDS half

// LDS K-stride in halves: 56 halves = 112 B = 7x16 B (keeps b128 alignment)
// and 28 DWORDs/row -> 16 consecutive rows hit distinct banks (28*r mod 64).
#define LSTR 56

// ---- CDNA5 feature probes (graceful fallback if a builtin is absent) ------
#if __has_builtin(__builtin_amdgcn_global_load_async_to_lds_b128)
#define HAS_ASYNC_LDS 1
#else
#define HAS_ASYNC_LDS 0
#endif
#if __has_builtin(__builtin_amdgcn_tensor_load_to_lds)
#define HAS_TDM 1
#else
#define HAS_TDM 0
#endif

__device__ __forceinline__ void wait_async_tensor() {
#if HAS_ASYNC_LDS
#if __has_builtin(__builtin_amdgcn_s_wait_asynccnt)
    __builtin_amdgcn_s_wait_asynccnt(0);
#else
    asm volatile("s_wait_asynccnt 0x0" ::: "memory");
#endif
#endif
#if HAS_TDM
    __builtin_amdgcn_s_wait_tensorcnt(0);
#endif
}

// Copy 16 B global -> LDS (async DMA path if available).
__device__ __forceinline__ void copy16_g2l(const _Float16* __restrict__ g,
                                           _Float16* l) {
#if HAS_ASYNC_LDS
    // Signature (from hipcc diagnostics): (int4 AS1*, int4 AS3*, imm, imm).
    __builtin_amdgcn_global_load_async_to_lds_b128((gas_v4i*)g, (las_v4i*)l, 0, 0);
#else
    *(v8h*)l = *(const v8h*)g;
#endif
}

#if HAS_TDM
// One TDM descriptor load: tile 128 rows x 32 halves from a row-major f16
// matrix with row stride KDIM halves, into LDS with +12-DWORD padding every
// 16 DWORDs (=> LDS row stride LSTR halves). Issued by one wave.
__device__ __forceinline__ void tdm_load_tile(const _Float16* __restrict__ g,
                                              unsigned lds_byte_addr) {
    const unsigned long long ga = (unsigned long long)g;
    u32x4 g0;
    g0[0] = 1u;                                        // count=1, user mode
    g0[1] = lds_byte_addr;                             // lds_addr [63:32]
    g0[2] = (unsigned)(ga & 0xFFFFFFFFull);            // global_addr lo
    g0[3] = (unsigned)((ga >> 32) & 0x01FFFFFFull)     // global_addr hi
          | (2u << 30);                                // type = 2 ("image")
    i32x8 g1;
    g1[0] = (1 << 16)      // data_size = 2 bytes
          | (1 << 20)      // pad_enable
          | (3 << 22)      // pad_interval: 16 DWORDs
          | (11 << 25);    // pad_amount: 12 DWORDs  -> row stride 56 halves
    g1[1] = (int)((unsigned)KDIM << 16);  // tensor_dim0 lo16 @ bits[63:48]
    g1[2] = (int)((unsigned)KDIM << 16);  // dim0 hi16 | tensor_dim1 lo16
    g1[3] = (int)(32u << 16);             // dim1 hi16 | tile_dim0 = 32
    g1[4] = 128;                          // tile_dim1 = 128 rows
    g1[5] = KDIM;                         // tensor_dim0_stride = 512 halves
    g1[6] = 0;
    g1[7] = 0;
    const i32x4 z4 = {0, 0, 0, 0};
    const i32x8 z8 = {0, 0, 0, 0, 0, 0, 0, 0};
    // 6-arg form (clang-23 / therock-10.0 toolchain).
    __builtin_amdgcn_tensor_load_to_lds(g0, g1, z4, z4, z8, 0);
}
#endif

// ---------------------------------------------------------------------------
// f32 -> f16 conversion (one-time, bandwidth-bound prologue).
// ---------------------------------------------------------------------------
__global__ __launch_bounds__(256)
void cvt_f32_f16(const float* __restrict__ src, _Float16* __restrict__ dst,
                 int n4) {
    const int i = blockIdx.x * 256 + threadIdx.x;
    if (i < n4) {
        const float4 v = ((const float4*)src)[i];
        v4h t; t[0] = (_Float16)v.x; t[1] = (_Float16)v.y;
               t[2] = (_Float16)v.z; t[3] = (_Float16)v.w;
        ((v4h*)dst)[i] = t;
    }
}

// ---------------------------------------------------------------------------
// GEMM: out[m, n] = sum_k A[m, k] * W[n, k] (+ bias). A, W are f16.
//   block = 256 threads (8 waves); block tile 128(M) x 128(N); K-step 32.
//   wave tile 32 x 64 = 2x4 accumulators of v_wmma_f32_16x16x32_f16.
//   Double-buffered LDS; A staged via async global->LDS b128 (ASYNCcnt),
//   W staged via one TDM tensor_load_to_lds per K-step (TENSORcnt).
// QKV=true : N = 1536 fused {Wq,Wk,Wv}; scatter to Q/K/V [B,H,S,DK] f32.
// QKV=false: N = 512; add bias; out row-major f32 [8192 x 512].
// ---------------------------------------------------------------------------
template <bool QKV>
__global__ __launch_bounds__(256)
void gemm_wmma_f16(const _Float16* __restrict__ A,
                   const _Float16* __restrict__ W0,
                   const _Float16* __restrict__ W1,
                   const _Float16* __restrict__ W2,
                   const float* __restrict__ bias,
                   float* __restrict__ outQ,
                   float* __restrict__ outK,
                   float* __restrict__ outV,
                   float* __restrict__ outP)
{
    __shared__ __align__(16) _Float16 lA[2][128][LSTR];
    __shared__ __align__(16) _Float16 lB[2][128][LSTR];

    const int tid   = threadIdx.x;
    const int lane  = tid & 31;
    const int wave  = tid >> 5;
    const int half  = lane >> 4;
    const int lrow  = lane & 15;
    const int waveM = wave & 3;    // 4 M-slots of 32 rows
    const int waveN = wave >> 2;   // 2 N-slots of 64 cols

    const int blockM = blockIdx.x * 128;
    const int nglob  = blockIdx.y * 128;
    const int wsel   = QKV ? (nglob >> 9) : 0;      // weight select (no straddle)
    const int nn0    = QKV ? (nglob & 511) : nglob; // column base within weight
    const _Float16* __restrict__ Wp =
        QKV ? (wsel == 0 ? W0 : (wsel == 1 ? W1 : W2)) : W0;

    // Stage A tile (128 x 32 halves): 512 16-B chunks over 256 threads.
    auto stageA = [&](int buf, int k0) {
        #pragma unroll
        for (int i = 0; i < 2; ++i) {
            const int c   = tid + i * 256;
            const int row = c >> 2;
            const int ko  = (c & 3) << 3;
            copy16_g2l(&A[(size_t)(blockM + row) * KDIM + k0 + ko],
                       &lA[buf][row][ko]);
        }
    };
    // Stage W tile (128 x 32 halves).
    auto stageB = [&](int buf, int k0) {
#if HAS_TDM
        if (wave == 0) {
            const unsigned laddr =
                (unsigned)(unsigned long long)(las_h*)&lB[buf][0][0];
            tdm_load_tile(&Wp[(size_t)nn0 * KDIM + k0], laddr);
        }
#else
        #pragma unroll
        for (int i = 0; i < 2; ++i) {
            const int c   = tid + i * 256;
            const int row = c >> 2;
            const int ko  = (c & 3) << 3;
            copy16_g2l(&Wp[(size_t)(nn0 + row) * KDIM + k0 + ko],
                       &lB[buf][row][ko]);
        }
#endif
    };

    v8f acc[2][4] = {};

    stageA(0, 0);
    stageB(0, 0);
    wait_async_tensor();
    __syncthreads();

    int buf = 0;
    for (int k0 = 0; k0 < KDIM; k0 += 32) {
        const int kn = k0 + 32;
        if (kn < KDIM) {               // prefetch next K-step into other buffer
            stageA(buf ^ 1, kn);
            stageB(buf ^ 1, kn);
        }

        // Fragments per the CDNA5 16-bit WMMA layouts.
        // A 16x32: lanes<16 hold K {0..7 | 16..23}; lanes>=16 {8..15 | 24..31}.
        v16h afr[2], bfr[4];
        #pragma unroll
        for (int mi = 0; mi < 2; ++mi) {
            const int arow = waveM * 32 + mi * 16 + lrow;
            const int kb   = half * 8;
            const v8h lo = *(const v8h*)&lA[buf][arow][kb];
            const v8h hi = *(const v8h*)&lA[buf][arow][kb + 16];
            #pragma unroll
            for (int e = 0; e < 8; ++e) { afr[mi][e] = lo[e]; afr[mi][e + 8] = hi[e]; }
        }
        // B 32x16: N = lane%16; lanes<16 hold K {0..7 | 16..23}; lanes>=16 {8..15 | 24..31}.
        #pragma unroll
        for (int ni = 0; ni < 4; ++ni) {
            const int brow = waveN * 64 + ni * 16 + lrow;
            const int kb   = half * 8;
            const v8h lo = *(const v8h*)&lB[buf][brow][kb];
            const v8h hi = *(const v8h*)&lB[buf][brow][kb + 16];
            #pragma unroll
            for (int e = 0; e < 8; ++e) { bfr[ni][e] = lo[e]; bfr[ni][e + 8] = hi[e]; }
        }

        #pragma unroll
        for (int mi = 0; mi < 2; ++mi)
            #pragma unroll
            for (int ni = 0; ni < 4; ++ni)
                acc[mi][ni] = __builtin_amdgcn_wmma_f32_16x16x32_f16(
                    false, afr[mi], false, bfr[ni],
                    (short)0, acc[mi][ni], false, false);

        wait_async_tensor();   // DMA of next tile overlapped with the WMMAs
        __syncthreads();
        buf ^= 1;
    }

    // Epilogue: C/D layout -> M = r + half*8, N = lane%16.
    #pragma unroll
    for (int mi = 0; mi < 2; ++mi) {
        #pragma unroll
        for (int ni = 0; ni < 4; ++ni) {
            const int cc    = nn0 + waveN * 64 + ni * 16 + lrow;
            const int rbase = blockM + waveM * 32 + mi * 16 + half * 8;
            const v8f c = acc[mi][ni];
            #pragma unroll
            for (int r = 0; r < 8; ++r) {
                const int row = rbase + r;
                const float val = c[r];
                if (QKV) {
                    const int b = row >> 12, s = row & (SEQ - 1);
                    const int h = cc >> 6,  d = cc & (DK - 1);
                    float* __restrict__ dst =
                        (wsel == 0) ? outQ : ((wsel == 1) ? outK : outV);
                    dst[((size_t)(b * NHEADS + h) * SEQ + s) * DK + d] = val;
                } else {
                    outP[(size_t)row * DMODEL + cc] = val + bias[cc];
                }
            }
        }
    }
}

// ---------------------------------------------------------------------------
// Banded attention, window +-4 (9 keys). One wave32 per query.
// Lanes 0..8 each own one key (full 64-dim dot, no cross-lane reduction);
// softmax via wave shfl max/sum; all 32 lanes accumulate P.V (2 dims each).
// Q/K/V: [B,H,S,DK] f32. Output: f16 rows [B*S, D] (feeds the WMMA GEMM).
// ---------------------------------------------------------------------------
__global__ __launch_bounds__(256)
void band_attn(const float* __restrict__ Q, const float* __restrict__ K,
               const float* __restrict__ V, _Float16* __restrict__ AO)
{
    const int lane = threadIdx.x & 31;
    const int wave = threadIdx.x >> 5;
    const int qid  = blockIdx.x * 8 + wave;     // enumerates [b][h][s]
    const int s    = qid & (SEQ - 1);
    const int bh   = qid >> 12;                 // b*8 + h
    const int h    = bh & (NHEADS - 1);
    const int b    = bh >> 3;

    const size_t qbase = (size_t)qid * DK;

    float score = -1e30f;
    if (lane < 9) {
        const int ks = s - 4 + lane;
        if (ks >= 0 && ks < SEQ) {
            const size_t kbase = qbase + (size_t)(ks - s) * DK;
            float acc = 0.f;
            #pragma unroll
            for (int d = 0; d < DK; ++d) acc += Q[qbase + d] * K[kbase + d];
            score = acc * 0.125f;               // d_k^-0.5, d_k = 64
        }
    }

    float m = score;
    #pragma unroll
    for (int off = 16; off > 0; off >>= 1) m = fmaxf(m, __shfl_xor(m, off, 32));
    const float p = __expf(score - m);          // 0 for masked / idle lanes
    float sum = p;
    #pragma unroll
    for (int off = 16; off > 0; off >>= 1) sum += __shfl_xor(sum, off, 32);
    const float inv = 1.0f / sum;

    float o0 = 0.f, o1 = 0.f;
    #pragma unroll
    for (int j = 0; j < 9; ++j) {
        const float pj = __shfl(p, j, 32) * inv;       // 0 when key invalid
        int ks = s - 4 + j;
        ks = ks < 0 ? 0 : (ks >= SEQ ? SEQ - 1 : ks);  // clamp: weight already 0
        const size_t vbase = qbase + (size_t)(ks - s) * DK;
        o0 += pj * V[vbase + lane];
        o1 += pj * V[vbase + lane + 32];
    }

    _Float16* __restrict__ orow = AO + (size_t)(b * SEQ + s) * DMODEL + h * DK;
    orow[lane]      = (_Float16)o0;
    orow[lane + 32] = (_Float16)o1;
}

// ---------------------------------------------------------------------------
// Pipeline:
//   cvt x,Wq,Wk,Wv,Wo -> f16     (one-time, bandwidth-bound)
//   QKV GEMM (WMMA, async+TDM staging) -> Q,K,V f32 [B,H,S,DK]
//   band_attn -> AO f16 [B*S, D]
//   output GEMM (WMMA) + bias -> out f32
// Workspace: Q|K|V (3 x 4,194,304 f32) then f16 region:
//   xh (4,194,304 h) | Wh[4] (4 x 262,144 h) | AOh (4,194,304 h)  ~= 66 MB,
// comfortably L2-resident (192 MB).
// ---------------------------------------------------------------------------
extern "C" void kernel_launch(void* const* d_in, const int* in_sizes, int n_in,
                              void* d_out, int out_size, void* d_ws, size_t ws_size,
                              hipStream_t stream)
{
    (void)in_sizes; (void)n_in; (void)out_size; (void)ws_size;

    const float* x  = (const float*)d_in[0];
    const float* Wq = (const float*)d_in[1];
    const float* Wk = (const float*)d_in[2];
    const float* Wv = (const float*)d_in[3];
    const float* Wo = (const float*)d_in[4];
    const float* bo = (const float*)d_in[5];
    float* out = (float*)d_out;

    const size_t QN = (size_t)BDIM * NHEADS * SEQ * DK;  // 4,194,304
    const size_t WN = (size_t)DMODEL * DMODEL;           // 262,144

    float* Qb = (float*)d_ws;
    float* Kb = Qb + QN;
    float* Vb = Kb + QN;
    _Float16* xh  = (_Float16*)(Vb + QN);
    _Float16* Wqh = xh  + QN;
    _Float16* Wkh = Wqh + WN;
    _Float16* Wvh = Wkh + WN;
    _Float16* Woh = Wvh + WN;
    _Float16* AOh = Woh + WN;

    const dim3 blk(256);

    // f32 -> f16 prologue (x and the four weights).
    cvt_f32_f16<<<dim3((int)(QN / 4 / 256)), blk, 0, stream>>>(x,  xh,  (int)(QN / 4));
    cvt_f32_f16<<<dim3((int)(WN / 4 / 256)), blk, 0, stream>>>(Wq, Wqh, (int)(WN / 4));
    cvt_f32_f16<<<dim3((int)(WN / 4 / 256)), blk, 0, stream>>>(Wk, Wkh, (int)(WN / 4));
    cvt_f32_f16<<<dim3((int)(WN / 4 / 256)), blk, 0, stream>>>(Wv, Wvh, (int)(WN / 4));
    cvt_f32_f16<<<dim3((int)(WN / 4 / 256)), blk, 0, stream>>>(Wo, Woh, (int)(WN / 4));

    // Fused QKV projection: M/128 = 64 blocks, fused N = 1536 -> 12 blocks.
    gemm_wmma_f16<true><<<dim3(MROWS / 128, (3 * DMODEL) / 128), blk, 0, stream>>>(
        xh, Wqh, Wkh, Wvh, nullptr, Qb, Kb, Vb, nullptr);

    // Banded softmax-attention: 65536 queries, 8 per block.
    band_attn<<<dim3((BDIM * NHEADS * SEQ) / 8), blk, 0, stream>>>(Qb, Kb, Vb, AOh);

    // Output projection + bias.
    gemm_wmma_f16<false><<<dim3(MROWS / 128, DMODEL / 128), blk, 0, stream>>>(
        AOh, Woh, nullptr, nullptr, bo, nullptr, nullptr, nullptr, out);
}
